// KeypointDetector_82489141887559
// MI455X (gfx1250) — compile-verified, matrix operand order checked by reading, not run on hardware
//
#include <hip/hip_runtime.h>
#include <hip/hip_bf16.h>
#include <math.h>

typedef __bf16 bf16;
typedef __attribute__((ext_vector_type(16))) __bf16 v16bf;
typedef __attribute__((ext_vector_type(8)))  __bf16 v8bf;
typedef __attribute__((ext_vector_type(8)))  float  v8f;

#define NB    8
#define CI    128
#define H0    120
#define W0    120
#define CM    256
#define CO    65
#define NP    80
#define KTOT  (CI*9)        // 1152
#define NPIX  (H0*W0)       // 14400
#define HS    (H0*8)        // 960
#define WSZ   (W0*8)        // 960
#define SPIX  (HS*WSZ)      // 921600
#define DET_T 0.0005f
#define CAP   4096
#define MAXK  1024

// ---------------- WMMA helpers ----------------
__device__ __forceinline__ v16bf frag_ld(const bf16* base, int row, int half) {
  // rows are 32 bf16; a lane's 16 values are two contiguous 16B chunks
  const v8bf lo = *(const v8bf*)(base + row * 32 + half * 8);
  const v8bf hi = *(const v8bf*)(base + row * 32 + 16 + half * 8);
  return __builtin_shufflevector(lo, hi, 0,1,2,3,4,5,6,7,8,9,10,11,12,13,14,15);
}
__device__ __forceinline__ v8f wmma_bf16(v16bf a, v16bf b, v8f c) {
  return __builtin_amdgcn_wmma_f32_16x16x32_bf16(false, a, false, b, (short)0, c, false, false);
}
// LDS byte offset of a __shared__ object (flat LDS addr low 32 bits == LDS offset)
__device__ __forceinline__ unsigned lds_off(const void* p) {
  return (unsigned)(unsigned long long)p;
}

// ---------------- layout conversion ----------------
__global__ __launch_bounds__(256) void cvt_x(const float* __restrict__ x, bf16* __restrict__ xb) {
  int idx = blockIdx.x * 256 + threadIdx.x;
  if (idx >= NB * NPIX * CI) return;
  int c = idx % CI, w = (idx / CI) % W0, h = (idx / (CI * W0)) % H0, b = idx / (CI * NPIX);
  xb[idx] = (bf16)x[(((size_t)b * CI + c) * H0 + h) * W0 + w];
}
__global__ __launch_bounds__(256) void cvt_wa(const float* __restrict__ wa, bf16* __restrict__ out) {
  int idx = blockIdx.x * 256 + threadIdx.x;   // [t][n][kr], t<36, n<256, kr<32
  if (idx >= KTOT * CM) return;
  int kr = idx % 32, n = (idx / 32) % CM, t = idx / (32 * CM);
  int k = t * 32 + kr;
  int khw = k >> 7, ci = k & 127;
  int kh = khw / 3, kw = khw % 3;
  out[idx] = (bf16)wa[(((size_t)n * CI + ci) * 3 + kh) * 3 + kw];
}
__global__ __launch_bounds__(256) void cvt_wb(const float* __restrict__ wb, bf16* __restrict__ out) {
  int idx = blockIdx.x * 256 + threadIdx.x;   // [t][n][kr], t<8, n<80, kr<32
  if (idx >= 8 * NP * 32) return;
  int kr = idx % 32, n = (idx / 32) % NP, t = idx / (32 * NP);
  int k = t * 32 + kr;
  out[idx] = (bf16)((n < CO) ? wb[(size_t)n * CM + k] : 0.0f);
}

// ---------------- conv 3x3 implicit GEMM (bf16 WMMA, async weight staging) ----------------
__global__ __launch_bounds__(256)
void conv3x3_wmma(const bf16* __restrict__ xb, const bf16* __restrict__ wa,
                  const float* __restrict__ ba, bf16* __restrict__ cpa) {
  const int nT = blockIdx.x;        // 0..1
  const int mT = blockIdx.y;        // 0..224
  const int b  = blockIdx.z;        // 0..7
  const int tid = threadIdx.x;
  const int lane = tid & 31, wave = tid >> 5;
  const int half = lane >> 4, lr = lane & 15;
  const int wm = wave & 1, wn = wave >> 1;

  __shared__ __align__(16) bf16 As[64 * 32];
  __shared__ __align__(16) bf16 Bs[128 * 32];     // [n][kr], 8KB contiguous image

  const int arow = tid >> 2;              // 0..63
  const int ac8  = (tid & 3) * 8;
  const int pix  = mT * 64 + arow;
  const int ph = pix / W0, pw = pix % W0;

  const unsigned bsOff  = lds_off(Bs) + (unsigned)tid * 16u;  // 16B per lane
  const unsigned bsGoff = (unsigned)tid * 16u;

  v8f acc[2][2] = {};

  for (int t = 0; t < 36; ++t) {
    __syncthreads();
    {   // stage B (128x32 bf16 == 8KB contiguous) via async global->LDS DMA
      const void* src = (const void*)(wa + ((size_t)(t * CM) + nT * 128) * 32);
      asm volatile(
        "global_load_async_to_lds_b128 %0, %1, %2\n\t"
        "global_load_async_to_lds_b128 %0, %1, %2 offset:4096"
        :: "v"(bsOff), "v"(bsGoff), "s"(src)
        : "memory");
    }
    {   // stage A (64x32) with implicit im2col + zero padding (needs predicated zero fill)
      const int khw = t >> 2;
      const int cb  = (t & 3) * 32;
      const int h2 = ph + khw / 3 - 1;
      const int w2 = pw + khw % 3 - 1;
      v8bf v = {};
      if (h2 >= 0 && h2 < H0 && w2 >= 0 && w2 < W0)
        v = *(const v8bf*)(xb + ((size_t)((b * H0 + h2) * W0 + w2)) * CI + cb + ac8);
      *(v8bf*)(As + arow * 32 + ac8) = v;
    }
    asm volatile("s_wait_asynccnt 0x0" ::: "memory");
    __syncthreads();
    v16bf a0 = frag_ld(As, wm * 32 + lr,      half);
    v16bf a1 = frag_ld(As, wm * 32 + 16 + lr, half);
    v16bf b0 = frag_ld(Bs, wn * 32 + lr,      half);
    v16bf b1 = frag_ld(Bs, wn * 32 + 16 + lr, half);
    acc[0][0] = wmma_bf16(a0, b0, acc[0][0]);
    acc[0][1] = wmma_bf16(a0, b1, acc[0][1]);
    acc[1][0] = wmma_bf16(a1, b0, acc[1][0]);
    acc[1][1] = wmma_bf16(a1, b1, acc[1][1]);
  }

  #pragma unroll
  for (int i = 0; i < 2; ++i)
    #pragma unroll
    for (int j = 0; j < 2; ++j) {
      const int n = nT * 128 + wn * 32 + j * 16 + lr;
      const float bias = ba[n];
      #pragma unroll
      for (int r = 0; r < 8; ++r) {
        const int m = mT * 64 + wm * 32 + i * 16 + r + half * 8;
        float v = acc[i][j][r] + bias;
        v = v > 0.f ? v : 0.f;
        cpa[((size_t)(b * NPIX + m)) * CM + n] = (bf16)v;
      }
    }
}

// ---------------- 1x1 conv GEMM + softmax + pixel-unshuffle ----------------
__global__ __launch_bounds__(128)
void conv1x1_softmax(const bf16* __restrict__ cpa, const bf16* __restrict__ wb,
                     const float* __restrict__ bb, float* __restrict__ smap) {
  const int mT = blockIdx.x;   // 0..224
  const int b  = blockIdx.y;
  const int tid = threadIdx.x;
  const int lane = tid & 31, wave = tid >> 5;
  const int half = lane >> 4, lr = lane & 15;

  __shared__ __align__(16) bf16  As[64 * 32];
  __shared__ __align__(16) bf16  Ws[NP * 32];     // [n][kr]
  __shared__ float Cs[64][NP];

  v8f acc[5] = {};
  const size_t pixBase = (size_t)b * NPIX + (size_t)mT * 64;

  // per-thread async A-tile descriptors: 256 chunks of 16B, 2 per thread
  // chunk c: row = c>>2, within = (c&3)*16 ; global row stride = CM*2 = 512B
  const unsigned aLds0 = lds_off(As) + (unsigned)tid * 16u;
  const unsigned aLds1 = aLds0 + 2048u;                          // +128 chunks
  const unsigned aGo0  = (unsigned)((tid >> 2) * (CM * 2) + (tid & 3) * 16);
  const unsigned aGo1  = aGo0 + 16384u;                          // +32 rows

  for (int t = 0; t < 8; ++t) {
    __syncthreads();
    {   // stage A (64x32) via async global->LDS DMA (always in-bounds, contiguous rows)
      const void* srcA = (const void*)(cpa + pixBase * CM + (size_t)t * 32);
      asm volatile(
        "global_load_async_to_lds_b128 %0, %2, %4\n\t"
        "global_load_async_to_lds_b128 %1, %3, %4"
        :: "v"(aLds0), "v"(aLds1), "v"(aGo0), "v"(aGo1), "s"(srcA)
        : "memory");
    }
    for (int c = tid; c < NP * 32 / 8; c += 128)
      *(v8bf*)(Ws + c * 8) = *(const v8bf*)(wb + (size_t)t * NP * 32 + c * 8);
    asm volatile("s_wait_asynccnt 0x0" ::: "memory");
    __syncthreads();
    v16bf a = frag_ld(As, wave * 16 + lr, half);
    #pragma unroll
    for (int j = 0; j < 5; ++j) {
      v16bf bfr = frag_ld(Ws, j * 16 + lr, half);
      acc[j] = wmma_bf16(a, bfr, acc[j]);
    }
  }

  #pragma unroll
  for (int j = 0; j < 5; ++j)
    #pragma unroll
    for (int r = 0; r < 8; ++r)
      Cs[wave * 16 + r + half * 8][j * 16 + lr] = acc[j][r];
  __syncthreads();

  if (tid < 64) {
    const int pixloc = mT * 64 + tid;
    const int ph = pixloc / W0, pw = pixloc % W0;
    float sc[CO];
    float mx = -1e30f;
    for (int c = 0; c < CO; ++c) { sc[c] = Cs[tid][c] + bb[c]; mx = sc[c] > mx ? sc[c] : mx; }
    float sum = 0.f;
    for (int c = 0; c < CO; ++c) { sc[c] = __expf(sc[c] - mx); sum += sc[c]; }
    const float inv = 1.f / sum;
    float* base = smap + (size_t)b * SPIX;
    for (int c = 0; c < 64; ++c) {
      const int y = ph * 8 + (c >> 3), x = pw * 8 + (c & 7);
      base[(size_t)y * WSZ + x] = sc[c] * inv;
    }
  }
}

// ---------------- NMS (separable 9x9 maxpool passes) ----------------
__global__ __launch_bounds__(256) void mp_h(const float* __restrict__ in, float* __restrict__ out) {
  int idx = blockIdx.x * 256 + threadIdx.x;
  if (idx >= NB * SPIX) return;
  int x = idx % WSZ;
  float m = -INFINITY;
  for (int dx = -4; dx <= 4; ++dx) {
    int xx = x + dx;
    if (xx >= 0 && xx < WSZ) m = fmaxf(m, in[idx - x + xx]);
  }
  out[idx] = m;
}
__global__ __launch_bounds__(256) void mp_h_ss(const float* __restrict__ P, const float* __restrict__ s,
                                               float* __restrict__ out) {
  int idx = blockIdx.x * 256 + threadIdx.x;
  if (idx >= NB * SPIX) return;
  int x = idx % WSZ;
  float m = -INFINITY;
  for (int dx = -4; dx <= 4; ++dx) {
    int xx = x + dx;
    if (xx >= 0 && xx < WSZ) {
      int j = idx - x + xx;
      float v = (P[j] > 0.f) ? 0.f : s[j];
      m = fmaxf(m, v);
    }
  }
  out[idx] = m;
}
__global__ __launch_bounds__(256) void mp_v(const float* __restrict__ in, float* __restrict__ out) {
  int idx = blockIdx.x * 256 + threadIdx.x;
  if (idx >= NB * SPIX) return;
  int y = (idx / WSZ) % HS;
  float m = -INFINITY;
  for (int dy = -4; dy <= 4; ++dy) {
    int yy = y + dy;
    if (yy >= 0 && yy < HS) m = fmaxf(m, in[idx + (yy - y) * WSZ]);
  }
  out[idx] = m;
}
__global__ __launch_bounds__(256) void nms_mask_init(const float* __restrict__ s, const float* __restrict__ P,
                                                     float* __restrict__ M) {
  int idx = blockIdx.x * 256 + threadIdx.x;
  if (idx >= NB * SPIX) return;
  M[idx] = (s[idx] == P[idx]) ? 1.f : 0.f;
}
__global__ __launch_bounds__(256) void nms_update(const float* __restrict__ P, const float* __restrict__ Q,
                                                  const float* __restrict__ s, float* __restrict__ M) {
  int idx = blockIdx.x * 256 + threadIdx.x;
  if (idx >= NB * SPIX) return;
  bool supp = P[idx] > 0.f;
  float ss = supp ? 0.f : s[idx];
  bool nm = (!supp) && (ss == Q[idx]);
  M[idx] = ((M[idx] != 0.f) || nm) ? 1.f : 0.f;
}
__global__ __launch_bounds__(256) void nms_finalize(const float* __restrict__ s, const float* __restrict__ M,
                                                    float* __restrict__ F) {
  int idx = blockIdx.x * 256 + threadIdx.x;
  if (idx >= NB * SPIX) return;
  int x = idx % WSZ;
  int y = (idx / WSZ) % HS;
  bool border = (x < 4) || (x >= WSZ - 4) || (y < 4) || (y >= HS - 4);
  float v = (M[idx] != 0.f) ? s[idx] : 0.f;
  F[idx] = (!border && v > DET_T) ? v : -1.f;
}

// ---------------- top-k ----------------
__global__ void zero_cnt(int* c) { if (threadIdx.x < NB) c[threadIdx.x] = 0; }

__global__ __launch_bounds__(256) void collect(const float* __restrict__ F, int* __restrict__ cnt,
                                               float* __restrict__ cval, int* __restrict__ cidx) {
  int idx = blockIdx.x * 256 + threadIdx.x;
  if (idx >= NB * SPIX) return;
  float v = F[idx];
  if (v > 0.f) {
    int b = idx / SPIX;
    int p = atomicAdd(&cnt[b], 1);
    if (p < CAP) { cval[b * CAP + p] = v; cidx[b * CAP + p] = idx - b * SPIX; }
  }
}

__global__ __launch_bounds__(1024)
void sort_emit(const int* __restrict__ cnt, const float* __restrict__ cval, const int* __restrict__ cidx,
               float* __restrict__ out) {
  __shared__ float sv[CAP];
  __shared__ int   si[CAP];
  const int b = blockIdx.x, tid = threadIdx.x;
  int n = cnt[b]; if (n > CAP) n = CAP;
  for (int t = tid; t < CAP; t += 1024) {
    sv[t] = (t < n) ? cval[b * CAP + t] : -1e30f;
    si[t] = (t < n) ? cidx[b * CAP + t] : 0;
  }
  __syncthreads();
  for (int k = 2; k <= CAP; k <<= 1)
    for (int j = k >> 1; j > 0; j >>= 1) {
      for (int t = tid; t < CAP; t += 1024) {
        int ixj = t ^ j;
        if (ixj > t) {
          bool dirDesc = ((t & k) == 0);
          float a = sv[t], c = sv[ixj];
          if ((a < c) == dirDesc) {
            sv[t] = c; sv[ixj] = a;
            int ti = si[t]; si[t] = si[ixj]; si[ixj] = ti;
          }
        }
      }
      __syncthreads();
    }
  if (tid < MAXK) {
    float v = -1.f; int x = 0, y = 0;
    if (tid < n) { v = sv[tid]; int id = si[tid]; x = id % WSZ; y = id / WSZ; }
    out[((size_t)b * MAXK + tid) * 2 + 0] = (float)x;
    out[((size_t)b * MAXK + tid) * 2 + 1] = (float)y;
    out[(size_t)NB * MAXK * 2 + (size_t)b * MAXK + tid] = v;
  }
}

// ---------------- driver ----------------
extern "C" void kernel_launch(void* const* d_in, const int* in_sizes, int n_in,
                              void* d_out, int out_size, void* d_ws, size_t ws_size,
                              hipStream_t stream) {
  (void)in_sizes; (void)n_in; (void)out_size; (void)ws_size;
  const float* x  = (const float*)d_in[0];
  const float* Wa = (const float*)d_in[1];
  const float* ba = (const float*)d_in[2];
  const float* Wb = (const float*)d_in[3];
  const float* bb = (const float*)d_in[4];
  float* out = (float*)d_out;

  char* ws = (char*)d_ws;
  size_t off = 0;
  auto carve = [&](size_t bytes) { char* p = ws + off; off = (off + bytes + 255) & ~(size_t)255; return p; };
  bf16*  xb   = (bf16*)carve((size_t)NB * NPIX * CI * sizeof(bf16));
  bf16*  wab  = (bf16*)carve((size_t)KTOT * CM * sizeof(bf16));
  bf16*  wbb  = (bf16*)carve((size_t)8 * NP * 32 * sizeof(bf16));
  bf16*  cpa  = (bf16*)carve((size_t)NB * NPIX * CM * sizeof(bf16));
  float* smap = (float*)carve((size_t)NB * SPIX * sizeof(float));
  float* T    = (float*)carve((size_t)NB * SPIX * sizeof(float));
  float* P    = (float*)carve((size_t)NB * SPIX * sizeof(float));
  float* Q    = (float*)carve((size_t)NB * SPIX * sizeof(float));
  float* M    = (float*)carve((size_t)NB * SPIX * sizeof(float));
  float* cval = (float*)carve((size_t)NB * CAP * sizeof(float));
  int*   cidx = (int*)carve((size_t)NB * CAP * sizeof(int));
  int*   cnt  = (int*)carve((size_t)NB * sizeof(int));

  const int ew = (NB * SPIX + 255) / 256;   // elementwise grid over score map

  cvt_x <<<(NB * NPIX * CI + 255) / 256, 256, 0, stream>>>(x, xb);
  cvt_wa<<<(KTOT * CM + 255) / 256,      256, 0, stream>>>(Wa, wab);
  cvt_wb<<<(8 * NP * 32 + 255) / 256,    256, 0, stream>>>(Wb, wbb);

  conv3x3_wmma<<<dim3(2, 225, NB), 256, 0, stream>>>(xb, wab, ba, cpa);
  conv1x1_softmax<<<dim3(225, NB), 128, 0, stream>>>(cpa, wbb, bb, smap);

  // simple_nms: initial max mask
  mp_h<<<ew, 256, 0, stream>>>(smap, T);
  mp_v<<<ew, 256, 0, stream>>>(T, P);
  nms_mask_init<<<ew, 256, 0, stream>>>(smap, P, M);
  // two suppression iterations
  for (int it = 0; it < 2; ++it) {
    mp_h<<<ew, 256, 0, stream>>>(M, T);
    mp_v<<<ew, 256, 0, stream>>>(T, P);                 // P = maxpool(max_mask)
    mp_h_ss<<<ew, 256, 0, stream>>>(P, smap, T);
    mp_v<<<ew, 256, 0, stream>>>(T, Q);                 // Q = maxpool(supp_scores)
    nms_update<<<ew, 256, 0, stream>>>(P, Q, smap, M);
  }
  nms_finalize<<<ew, 256, 0, stream>>>(smap, M, T);     // T now holds masked scores

  zero_cnt<<<1, 32, 0, stream>>>(cnt);
  collect<<<ew, 256, 0, stream>>>(T, cnt, cval, cidx);
  sort_emit<<<NB, 1024, 0, stream>>>(cnt, cval, cidx, out);
}
